// MoSWithSharedEmbedding_57724360458762
// MI455X (gfx1250) — matrix operand order repaired and checked by reading, not compiled, not run on hardware
//
#include <hip/hip_runtime.h>
#include <hip/hip_bf16.h>
#include <cstdint>

typedef __bf16 bf16;
typedef __bf16 v16bf __attribute__((ext_vector_type(16)));
typedef __bf16 v8bf  __attribute__((ext_vector_type(8)));
typedef float  v8f   __attribute__((ext_vector_type(8)));
typedef int    v4i   __attribute__((ext_vector_type(4)));

#define S_DIM 256
#define H_DIM 2560
#define E_DIM 10
#define V_DIM 50257
#define VPAD  50304   // 786 * 64

// Async global->LDS path (CDNA5): gated so a missing builtin falls back to the
// synchronous staging path.
#if defined(__AMDGCN__) && __has_builtin(__builtin_amdgcn_global_load_async_to_lds_b128) && __has_builtin(__builtin_amdgcn_s_wait_asynccnt)
#define MOS_ASYNC 1
#else
#define MOS_ASYNC 0
#endif

#if MOS_ASYNC
// Builtin signature (from clang diagnostic): (v4i AS1*, v4i AS3*, imm, imm).
// Use integer round-trips for the address-space conversions: global generic VA
// == AS1 VA; for LDS the low 32 bits of a generic aperture address are the LDS address (ISA
// 10.2), so truncating to 32 bits yields the AS3 offset.
__device__ inline void async_cp16(const void* g, void* l) {
  __attribute__((address_space(1))) v4i* gp =
      (__attribute__((address_space(1))) v4i*)(uintptr_t)g;
  __attribute__((address_space(3))) v4i* lp =
      (__attribute__((address_space(3))) v4i*)(uint32_t)(uintptr_t)l;
  __builtin_amdgcn_global_load_async_to_lds_b128(gp, lp, /*offset=*/0,
                                                 /*cpol=*/0);
}
#endif

// ---------------- WMMA helpers (CDNA5 16x16x32 bf16, f32 accum) ----------------

__device__ inline v8f v8f_zero() {
  v8f z; for (int i = 0; i < 8; ++i) z[i] = 0.0f; return z;
}

// A fragment: 16x32 bf16. Lane (row = lane&15) holds K-window offsets
// [0..15] (lanes 0-15) or [16..31] (lanes 16-31).
__device__ inline v16bf load_a_frag(const bf16* rowp, int ks, int lane) {
  const int kb = ks + ((lane & 16) ? 16 : 0);
  union { v16bf v; v8bf h[2]; } u;
  u.h[0] = *(const v8bf*)(rowp + kb);
  u.h[1] = *(const v8bf*)(rowp + kb + 8);
  return u.v;
}

// B fragment: 32x16 bf16 from n-major LDS tile Bt[n][64]. Lane (col = lane&15)
// holds K = kb..kb+15 contiguous (kb = ks, or ks+16 for lanes 16-31).
__device__ inline v16bf load_b_frag(const bf16* Bt, int n, int ks, int lane) {
  const int kb = ks + ((lane & 16) ? 16 : 0);
  return *(const v16bf*)(Bt + (size_t)(n + (lane & 15)) * 64 + kb);
}

__device__ inline v8f wmma_bf16f32(v16bf a, v16bf b, v8f c) {
  return __builtin_amdgcn_wmma_f32_16x16x32_bf16(false, a, false, b,
                                                 (short)0, c, false, false);
}

__device__ inline uint32_t pack_bf16(float lo, float hi) {
  union { bf16 h[2]; uint32_t u; } p;
  p.h[0] = (bf16)lo;
  p.h[1] = (bf16)hi;
  return p.u;
}

// ---------------- K1: RMSNorm + bf16 h + gate softmax ----------------

__global__ __launch_bounds__(256) void rmsnorm_gate_kernel(
    const float* __restrict__ x, const float* __restrict__ scale,
    const float* __restrict__ gw, bf16* __restrict__ hout,
    float* __restrict__ gate_prob)
{
  const int s = blockIdx.x;
  const int t = threadIdx.x;
  __shared__ float red[256];
  __shared__ float gred[E_DIM][256];
  __shared__ float glog[E_DIM];

  float xv[10];
  float ss = 0.f;
  for (int i = 0; i < 10; ++i) {
    const int hh = t + i * 256;
    xv[i] = x[(size_t)s * H_DIM + hh];
    ss += xv[i] * xv[i];
  }
  red[t] = ss;
  __syncthreads();
  for (int off = 128; off > 0; off >>= 1) {
    if (t < off) red[t] += red[t + off];
    __syncthreads();
  }
  const float rinv = rsqrtf(red[0] / (float)H_DIM + 1e-5f);

  float gpart[E_DIM];
  for (int e = 0; e < E_DIM; ++e) gpart[e] = 0.f;
  for (int i = 0; i < 10; ++i) {
    const int hh = t + i * 256;
    const float hn = xv[i] * rinv * scale[hh];
    hout[(size_t)s * H_DIM + hh] = (bf16)hn;
    for (int e = 0; e < E_DIM; ++e) gpart[e] += hn * gw[(size_t)hh * E_DIM + e];
  }
  for (int e = 0; e < E_DIM; ++e) gred[e][t] = gpart[e];
  __syncthreads();
  if (t < E_DIM) {
    float acc = 0.f;
    for (int i = 0; i < 256; ++i) acc += gred[t][i];
    glog[t] = acc;
  }
  __syncthreads();
  if (t == 0) {
    float m = glog[0];
    for (int e = 1; e < E_DIM; ++e) m = fmaxf(m, glog[e]);
    float sum = 0.f, ex[E_DIM];
    for (int e = 0; e < E_DIM; ++e) { ex[e] = __expf(glog[e] - m); sum += ex[e]; }
    for (int e = 0; e < E_DIM; ++e) gate_prob[s * E_DIM + e] = ex[e] / sum;
  }
}

// ---------------- K2: embedding f32 -> bf16 (tail rows zero-padded) ----------------

__global__ void convert_emb_kernel(const float* __restrict__ emb,
                                   bf16* __restrict__ out)
{
  const size_t total4 = (size_t)VPAD * H_DIM / 4;
  const size_t valid4 = (size_t)V_DIM * H_DIM / 4;
  for (size_t i = blockIdx.x * (size_t)blockDim.x + threadIdx.x; i < total4;
       i += (size_t)gridDim.x * blockDim.x) {
    uint32_t lo, hi;
    if (i < valid4) {
      const float4 f = *(const float4*)(emb + i * 4);
      lo = pack_bf16(f.x, f.y);
      hi = pack_bf16(f.z, f.w);
    } else {
      lo = 0u; hi = 0u;
    }
    uint32_t* o32 = (uint32_t*)(out + i * 4);   // 8-byte aligned
    o32[0] = lo;
    o32[1] = hi;
  }
}

// ---------------- K3: expert_hidden = h @ W_e  (bf16 WMMA, f32 accum) ----------------
// eh rows r = e*256 + s, cols d (H).  Block = 128 thr (4 waves), tile 32x64.

__global__ __launch_bounds__(128) void expert_gemm_kernel(
    const bf16* __restrict__ h, const float* __restrict__ W,
    bf16* __restrict__ eh)
{
  __shared__ alignas(32) bf16 As[32 * 64];
  __shared__ alignas(32) bf16 Bt[64 * 64];
  const int t = threadIdx.x;
  const int lane = t & 31;
  const int wv = t >> 5;
  const int n0 = blockIdx.x * 64;
  const int s0 = blockIdx.y * 32;
  const int e  = blockIdx.z;
  const int m0w = (wv & 1) * 16;
  const int n0w = (wv >> 1) * 32;
  v8f acc0 = v8f_zero(), acc1 = v8f_zero();
  const size_t Wbase = (size_t)e * H_DIM * H_DIM;
  uint32_t* bt32 = (uint32_t*)Bt;

  for (int k0 = 0; k0 < H_DIM; k0 += 64) {
    // A tile 32x64 bf16 (row-major, k contiguous)
    for (int it = 0; it < 2; ++it) {
      const int c = it * 128 + t;           // 256 chunks of 8 bf16
      const int m = c >> 3, kk = (c & 7) * 8;
      *(v8bf*)(As + m * 64 + kk) =
          *(const v8bf*)(h + (size_t)(s0 + m) * H_DIM + k0 + kk);
    }
    // B tile: W[e][k][n] f32 -> transposed bf16 Bt[n][k]; two k-rows at a time
    // so LDS writes are packed b32 (k is the contiguous axis of Bt).
    for (int it = 0; it < 4; ++it) {
      const int c = it * 128 + t;           // 512 pair-chunks
      const int kp = (c >> 4) * 2;          // even k
      const int n4 = (c & 15) * 4;
      const float4 f0 =
          *(const float4*)(W + Wbase + (size_t)(k0 + kp) * H_DIM + n0 + n4);
      const float4 f1 =
          *(const float4*)(W + Wbase + (size_t)(k0 + kp + 1) * H_DIM + n0 + n4);
      bt32[((n4 + 0) * 64 + kp) >> 1] = pack_bf16(f0.x, f1.x);
      bt32[((n4 + 1) * 64 + kp) >> 1] = pack_bf16(f0.y, f1.y);
      bt32[((n4 + 2) * 64 + kp) >> 1] = pack_bf16(f0.z, f1.z);
      bt32[((n4 + 3) * 64 + kp) >> 1] = pack_bf16(f0.w, f1.w);
    }
    __syncthreads();
    const bf16* arow = As + (size_t)(m0w + (lane & 15)) * 64;
    for (int ks = 0; ks < 64; ks += 32) {
      const v16bf a  = load_a_frag(arow, ks, lane);
      const v16bf b0 = load_b_frag(Bt, n0w,      ks, lane);
      const v16bf b1 = load_b_frag(Bt, n0w + 16, ks, lane);
      acc0 = wmma_bf16f32(a, b0, acc0);
      acc1 = wmma_bf16f32(a, b1, acc1);
    }
    __syncthreads();
  }
  const int colbase = n0 + n0w + (lane & 15);
  const int mbase = m0w + ((lane & 16) ? 8 : 0);
  for (int i = 0; i < 8; ++i) {
    const int r = e * S_DIM + s0 + mbase + i;
    eh[(size_t)r * H_DIM + colbase]      = (bf16)acc0[i];
    eh[(size_t)r * H_DIM + colbase + 16] = (bf16)acc1[i];
  }
}

// ---------------- K4/K6: the big fused GEMM over vocab ----------------
// Block = 320 thr = 10 waves (wave == expert), tile 16 s x 64 v, K = 2560.
// Async path: double-buffered LDS, GLOBAL_LOAD_ASYNC_TO_LDS_B128 streams the
// next K-chunk while WMMAs consume the current one (ASYNCcnt-tracked).
// MODE 0: accumulate Z[s,e] = sum_v exp(logit) via LDS + global atomics.
// MODE 1: mixed[s,v] = sum_e wgate[s,e]*exp(logit); out = log(mixed + 1e-10).

template <int MODE>
__global__ __launch_bounds__(320) void mos_logits_kernel(
    const bf16* __restrict__ eh, const bf16* __restrict__ emb,
    const float* __restrict__ wgate, float* __restrict__ Z,
    float* __restrict__ out)
{
#if MOS_ASYNC
  __shared__ alignas(32) bf16 As[2][160 * 64];
  __shared__ alignas(32) bf16 Bt[2][64 * 64];
#else
  __shared__ alignas(32) bf16 As[1][160 * 64];
  __shared__ alignas(32) bf16 Bt[1][64 * 64];
#endif
  __shared__ float mix[16 * 64];
  __shared__ float rowacc[160];
  const int t = threadIdx.x;
  const int lane = t & 31;
  const int e = t >> 5;
  const int v0 = blockIdx.x * 64;
  const int s0 = blockIdx.y * 16;

  if (MODE == 0) {
    if (t < 160) rowacc[t] = 0.f;
  } else {
    for (int i = t; i < 1024; i += 320) mix[i] = 0.f;
  }

  // Per-thread staging assignments (constant across K-chunks).
  // A: 160 rows x 64, row a = e*16 + m -> global row r = e*256 + s0 + m.
  size_t a_goff[4];
  int a_loff[4];
  for (int it = 0; it < 4; ++it) {
    const int c = it * 320 + t;            // 1280 chunks of 8 bf16
    const int a = c >> 3, kk = (c & 7) * 8;
    const int r = (a >> 4) * S_DIM + s0 + (a & 15);
    a_goff[it] = (size_t)r * H_DIM + kk;
    a_loff[it] = a * 64 + kk;
  }
  // B: 64 rows x 64, direct copy (emb is [v][k], already n-major/k-contig).
  const int bn0 = t >> 3, bk0 = (t & 7) * 8;          // chunks 0..319
  const size_t b_goff0 = (size_t)(v0 + bn0) * H_DIM + bk0;
  const int b_loff0 = bn0 * 64 + bk0;
  const bool have_b1 = (t < 192);                      // chunks 320..511
  const int c2 = 320 + t;
  const int bn1 = c2 >> 3, bk1 = (c2 & 7) * 8;
  const size_t b_goff1 = (size_t)(v0 + bn1) * H_DIM + bk1;
  const int b_loff1 = bn1 * 64 + bk1;

  v8f acc[4];
  for (int j = 0; j < 4; ++j) acc[j] = v8f_zero();

  const int NCH = H_DIM / 64;

#if MOS_ASYNC
  auto issue = [&](int buf, int k0) {
    for (int it = 0; it < 4; ++it)
      async_cp16(eh + a_goff[it] + k0, &As[buf][a_loff[it]]);
    async_cp16(emb + b_goff0 + k0, &Bt[buf][b_loff0]);
    if (have_b1)
      async_cp16(emb + b_goff1 + k0, &Bt[buf][b_loff1]);
  };
  issue(0, 0);
  for (int ci = 0; ci < NCH; ++ci) {
    const int cur = ci & 1;
    if (ci + 1 < NCH) {
      issue(cur ^ 1, (ci + 1) * 64);
      // Async loads complete in order: <=5 outstanding (of the >=5 just
      // issued) implies the previous chunk fully landed.
      __builtin_amdgcn_s_wait_asynccnt(5);
    } else {
      __builtin_amdgcn_s_wait_asynccnt(0);
    }
    __syncthreads();
    const bf16* arow = &As[cur][(e * 16 + (lane & 15)) * 64];
    for (int ks = 0; ks < 64; ks += 32) {
      const v16bf a = load_a_frag(arow, ks, lane);
      for (int j = 0; j < 4; ++j)
        acc[j] = wmma_bf16f32(a, load_b_frag(&Bt[cur][0], j * 16, ks, lane),
                              acc[j]);
    }
    __syncthreads();
  }
#else
  for (int ci = 0; ci < NCH; ++ci) {
    const int k0 = ci * 64;
    for (int it = 0; it < 4; ++it)
      *(v8bf*)(&As[0][a_loff[it]]) = *(const v8bf*)(eh + a_goff[it] + k0);
    *(v8bf*)(&Bt[0][b_loff0]) = *(const v8bf*)(emb + b_goff0 + k0);
    if (k0 + 64 < H_DIM)
      __builtin_prefetch(emb + b_goff0 + k0 + 64, 0, 0);
    if (have_b1)
      *(v8bf*)(&Bt[0][b_loff1]) = *(const v8bf*)(emb + b_goff1 + k0);
    __syncthreads();
    const bf16* arow = &As[0][(e * 16 + (lane & 15)) * 64];
    for (int ks = 0; ks < 64; ks += 32) {
      const v16bf a = load_a_frag(arow, ks, lane);
      for (int j = 0; j < 4; ++j)
        acc[j] = wmma_bf16f32(a, load_b_frag(&Bt[0][0], j * 16, ks, lane),
                              acc[j]);
    }
    __syncthreads();
  }
#endif

  const int colb = lane & 15;
  const int mbase = (lane & 16) ? 8 : 0;
  if (MODE == 0) {
    for (int i = 0; i < 8; ++i) {
      float rs = 0.f;
      for (int j = 0; j < 4; ++j) {
        const int v = v0 + j * 16 + colb;
        if (v < V_DIM) rs += __expf(acc[j][i]);
      }
      atomicAdd(&rowacc[e * 16 + mbase + i], rs);   // ds_add_f32
    }
    __syncthreads();
    if (t < 160) {
      const int r = (t >> 4) * S_DIM + s0 + (t & 15);
      atomicAdd(&Z[r], rowacc[t]);                  // global_atomic_add_f32
    }
  } else {
    float wrow[8];
    for (int i = 0; i < 8; ++i)
      wrow[i] = wgate[e * S_DIM + s0 + mbase + i];
    for (int j = 0; j < 4; ++j) {
      const int v = v0 + j * 16 + colb;
      if (v < V_DIM) {
        for (int i = 0; i < 8; ++i)
          atomicAdd(&mix[(mbase + i) * 64 + j * 16 + colb],
                    wrow[i] * __expf(acc[j][i]));
      }
    }
    __syncthreads();
    for (int idx = t; idx < 1024; idx += 320) {
      const int m = idx >> 6, c = idx & 63;
      const int v = v0 + c;
      if (v < V_DIM)
        out[(size_t)(s0 + m) * V_DIM + v] = __logf(mix[idx] + 1e-10f);
    }
  }
}

// ---------------- K5: wgate[s,e] = gate_prob / Z ----------------

__global__ void gate_over_z_kernel(const float* __restrict__ gate_prob,
                                   const float* __restrict__ Z,
                                   float* __restrict__ wg)
{
  const int r = blockIdx.x * blockDim.x + threadIdx.x;
  if (r < E_DIM * S_DIM) {
    const int e = r / S_DIM, s = r % S_DIM;
    wg[r] = gate_prob[s * E_DIM + e] / fmaxf(Z[r], 1e-30f);
  }
}

// ---------------- launch ----------------

extern "C" void kernel_launch(void* const* d_in, const int* in_sizes, int n_in,
                              void* d_out, int out_size, void* d_ws, size_t ws_size,
                              hipStream_t stream) {
  (void)in_sizes; (void)n_in; (void)out_size; (void)ws_size;
  const float* x     = (const float*)d_in[0];   // hidden_states (1,256,2560)
  const float* embf  = (const float*)d_in[1];   // embedding (50257,2560)
  const float* scale = (const float*)d_in[2];   // norm_scale (2560,)
  const float* W     = (const float*)d_in[3];   // expert_weights (10,2560,2560)
  const float* gw    = (const float*)d_in[4];   // gate_weight (2560,10)
  float* out = (float*)d_out;                   // (1,256,50257) f32

  char* ws = (char*)d_ws;
  size_t off = 0;
  auto carve = [&](size_t bytes) -> void* {
    void* p = ws + off;
    off += bytes;
    off = (off + 255) & ~(size_t)255;
    return p;
  };
  bf16*  h_bf   = (bf16*) carve((size_t)S_DIM * H_DIM * 2);           // 1.25 MB
  bf16*  eh     = (bf16*) carve((size_t)E_DIM * S_DIM * H_DIM * 2);   // 12.5 MB
  float* gate_p = (float*)carve((size_t)S_DIM * E_DIM * 4);
  float* Zp     = (float*)carve((size_t)E_DIM * S_DIM * 4);
  float* wg     = (float*)carve((size_t)E_DIM * S_DIM * 4);
  bf16*  emb_bf = (bf16*) carve((size_t)VPAD * H_DIM * 2);            // 246 MB

  (void)hipMemsetAsync(Zp, 0, (size_t)E_DIM * S_DIM * 4, stream);
  convert_emb_kernel<<<2048, 256, 0, stream>>>(embf, emb_bf);
  rmsnorm_gate_kernel<<<S_DIM, 256, 0, stream>>>(x, scale, gw, h_bf, gate_p);
  expert_gemm_kernel<<<dim3(H_DIM / 64, S_DIM / 32, E_DIM), 128, 0, stream>>>(
      h_bf, W, eh);
  mos_logits_kernel<0><<<dim3(VPAD / 64, S_DIM / 16), 320, 0, stream>>>(
      eh, emb_bf, nullptr, Zp, nullptr);
  gate_over_z_kernel<<<(E_DIM * S_DIM + 255) / 256, 256, 0, stream>>>(
      gate_p, Zp, wg);
  mos_logits_kernel<1><<<dim3(VPAD / 64, S_DIM / 16), 320, 0, stream>>>(
      eh, emb_bf, wg, nullptr, out);
}
